// GNN_44143673868930
// MI455X (gfx1250) — matrix operand order repaired and checked by reading, not compiled, not run on hardware
//
#include <hip/hip_runtime.h>

// ---------------------------------------------------------------------------
// GraphNet GNN on MI455X (gfx1250): all GEMMs via v_wmma_f32_16x16x32_f16.
// One wave32 processes one 16-row tile through a full 3-layer MLP.
// ---------------------------------------------------------------------------

#define LN_EPS 1e-5f
#define WAVES 4          // waves (tiles) per block -> 128 threads
#define HID 64

typedef _Float16 h16 __attribute__((ext_vector_type(16)));
typedef _Float16 h8v __attribute__((ext_vector_type(8)));
typedef float    f8  __attribute__((ext_vector_type(8)));

#define MEMFENCE() asm volatile("" ::: "memory")

struct MLPW {
  const _Float16 *w0, *w1, *w2;   // packed B-fragment weights
  const float *b0, *b1, *b2;      // biases (f32)
  const float *gamma, *beta;      // layernorm params
};

__device__ __forceinline__ f8 wmma16(h16 a, h16 b, f8 c) {
  return __builtin_amdgcn_wmma_f32_16x16x32_f16(false, a, false, b, (short)0, c,
                                                false, false);
}

// A fragment from LDS f16 tile (row-major, strideH halfs/row).
// 16-bit A 16x32 (contiguous, matches packed B): lanes 0-15 rows M,
// elems 0..15 -> K k0..k0+15, with k0 = 16*(lane>=16) + 32*chunk.
__device__ __forceinline__ h16 load_a_lds(const _Float16* tile, int strideH,
                                          int lane, int chunk) {
  int row = lane & 15, hf = lane >> 4;
  const _Float16* p = tile + row * strideH + chunk * 32 + hf * 16;
  h8v lo = *(const h8v*)p;
  h8v hi = *(const h8v*)(p + 8);
  return __builtin_shufflevector(lo, hi, 0, 1, 2, 3, 4, 5, 6, 7, 8, 9, 10, 11,
                                 12, 13, 14, 15);
}

template <int CHUNKS>
__device__ __forceinline__ void gemm_lds(const _Float16* tile, int strideH,
                                         const _Float16* wp, int lane, f8 c[4]) {
#pragma unroll
  for (int k = 0; k < CHUNKS; ++k) {
    h16 a = load_a_lds(tile, strideH, lane, k);
#pragma unroll
    for (int n = 0; n < 4; ++n) {
      h16 b = *(const h16*)(wp + (size_t)(k * 4 + n) * 512 + lane * 16);
      c[n] = wmma16(a, b, c[n]);
    }
  }
}

__device__ __forceinline__ void bias_act(f8 c[4], const float* bias, int lane,
                                         bool relu) {
  int col = lane & 15;
#pragma unroll
  for (int n = 0; n < 4; ++n) {
    float bv = bias[n * 16 + col];
#pragma unroll
    for (int r = 0; r < 8; ++r) {
      float v = c[n][r] + bv;
      c[n][r] = relu ? fmaxf(v, 0.f) : v;
    }
  }
}

// C layout: lane holds col (lane&15) of tile n; rows r + 8*(lane>=16) in VGPR r
__device__ __forceinline__ void c_to_lds_h(const f8 c[4], _Float16* tile,
                                           int strideH, int lane) {
  int col = lane & 15, hf = lane >> 4;
#pragma unroll
  for (int n = 0; n < 4; ++n)
#pragma unroll
    for (int r = 0; r < 8; ++r)
      tile[(r + hf * 8) * strideH + n * 16 + col] = (_Float16)c[n][r];
}

// Register-only LayerNorm: sum 4 frags per VGPR slot, reduce across the
// 16-lane half with shfl_xor(1,2,4,8). Stats are then uniform per half-row.
__device__ __forceinline__ void layernorm(f8 c[4], const float* gamma,
                                          const float* beta, int lane) {
  float s1[8], s2[8];
#pragma unroll
  for (int r = 0; r < 8; ++r) {
    s1[r] = c[0][r] + c[1][r] + c[2][r] + c[3][r];
    s2[r] = c[0][r] * c[0][r] + c[1][r] * c[1][r] + c[2][r] * c[2][r] +
            c[3][r] * c[3][r];
  }
#pragma unroll
  for (int m = 1; m <= 8; m <<= 1) {
#pragma unroll
    for (int r = 0; r < 8; ++r) {
      s1[r] += __shfl_xor(s1[r], m, 32);
      s2[r] += __shfl_xor(s2[r], m, 32);
    }
  }
  int col = lane & 15;
  float g[4], bt[4];
#pragma unroll
  for (int n = 0; n < 4; ++n) {
    g[n] = gamma[n * 16 + col];
    bt[n] = beta[n * 16 + col];
  }
#pragma unroll
  for (int r = 0; r < 8; ++r) {
    float mu = s1[r] * (1.f / 64.f);
    float var = s2[r] * (1.f / 64.f) - mu * mu;
    float rs = rsqrtf(var + LN_EPS);
#pragma unroll
    for (int n = 0; n < 4; ++n) c[n][r] = (c[n][r] - mu) * rs * g[n] + bt[n];
  }
}

// 32 consecutive f32 from global -> f16 into LDS (coalesced b128 loads)
__device__ __forceinline__ void stage32(const float* g, _Float16* l) {
  const float4* gv = (const float4*)g;
#pragma unroll
  for (int j = 0; j < 8; ++j) {
    float4 v = gv[j];
    l[j * 4 + 0] = (_Float16)v.x;
    l[j * 4 + 1] = (_Float16)v.y;
    l[j * 4 + 2] = (_Float16)v.z;
    l[j * 4 + 3] = (_Float16)v.w;
  }
}

// ---------------------------------------------------------------------------
// Weight packing: f32 [K,64] (row-major, padded K) -> per-fragment f16 B tiles.
// Packed B 32x16 (16-bit, this code's contiguous k): lanes 0-15 hold K k0..k0+15 consecutively,
// lanes 16-31 hold k0+16..k0+31; lane&15 = column. 512 halfs per fragment.
// ---------------------------------------------------------------------------
struct PackArgs {
  const float* W[32];
  int Kpad[32];
  int Kreal[32];
  int off[32];  // in halfs
};

__global__ void pack_kernel(PackArgs pa, _Float16* __restrict__ wp) {
  int f = blockIdx.x;
  int lane = threadIdx.x;
  int m = -1, lf = 0, fb = 0;
  for (int i = 0; i < 32; ++i) {
    int nf = (pa.Kpad[i] >> 5) * 4;
    if (m < 0 && f < fb + nf) { m = i; lf = f - fb; }
    fb += nf;
  }
  int chunk = lf >> 2, nt = lf & 3;
  int col = nt * 16 + (lane & 15);
  int kb = chunk * 32 + (lane >> 4) * 16;
  _Float16* dst = wp + pa.off[m] + (size_t)lf * 512 + lane * 16;
  const float* W = pa.W[m];
  int Kr = pa.Kreal[m];
  for (int i = 0; i < 16; ++i) {
    int K = kb + i;
    dst[i] = (_Float16)((K < Kr) ? W[(size_t)K * 64 + col] : 0.f);
  }
}

// ---------------------------------------------------------------------------
// Encoder: feat[N,4] -> out[N,64] through 4->64->64->64 MLP + LN.
// ---------------------------------------------------------------------------
__global__ void __launch_bounds__(WAVES * 32)
encoder_kernel(const float* __restrict__ feat, float* __restrict__ xout,
               MLPW w, int nRows) {
  __shared__ _Float16 act[WAVES][16 * 72];
  __shared__ float obuf[WAVES][16 * 68];
  int lane = threadIdx.x & 31, wave = threadIdx.x >> 5;
  int tile = blockIdx.x * WAVES + wave;
  if (tile * 16 >= nRows) return;
  int row = lane & 15, hf = lane >> 4;
  int grow = tile * 16 + row;

  // layer 0: K=4 (padded to 32); build A directly, zeros beyond K=3
  h16 a0 = (h16)(_Float16)0;
  if (hf == 0) {
    float4 v = *(const float4*)(feat + (size_t)grow * 4);
    a0[0] = (_Float16)v.x; a0[1] = (_Float16)v.y;
    a0[2] = (_Float16)v.z; a0[3] = (_Float16)v.w;
  }
  f8 c[4];
#pragma unroll
  for (int n = 0; n < 4; ++n) c[n] = (f8)0.f;
#pragma unroll
  for (int n = 0; n < 4; ++n) {
    h16 b = *(const h16*)(w.w0 + (size_t)n * 512 + lane * 16);
    c[n] = wmma16(a0, b, c[n]);
  }
  bias_act(c, w.b0, lane, true);
  c_to_lds_h(c, act[wave], 72, lane);

#pragma unroll
  for (int n = 0; n < 4; ++n) c[n] = (f8)0.f;
  gemm_lds<2>(act[wave], 72, w.w1, lane, c);
  bias_act(c, w.b1, lane, true);
  c_to_lds_h(c, act[wave], 72, lane);

#pragma unroll
  for (int n = 0; n < 4; ++n) c[n] = (f8)0.f;
  gemm_lds<2>(act[wave], 72, w.w2, lane, c);
  bias_act(c, w.b2, lane, false);
  layernorm(c, w.gamma, w.beta, lane);

  // coalesced f32 writeback via LDS
  int col = lane & 15;
#pragma unroll
  for (int n = 0; n < 4; ++n)
#pragma unroll
    for (int r = 0; r < 8; ++r)
      obuf[wave][(r + hf * 8) * 68 + n * 16 + col] = c[n][r];
  MEMFENCE();
  const float* src = &obuf[wave][row * 68 + hf * 32];
  float* dst = xout + (size_t)grow * HID + hf * 32;
#pragma unroll
  for (int j = 0; j < 8; ++j) ((float4*)dst)[j] = ((const float4*)src)[j];
}

// ---------------------------------------------------------------------------
// MP edge pass: m = MLP([x[dst]|x[src]|e]); e += m; aggr[dst] += m (atomics).
// ---------------------------------------------------------------------------
__global__ void __launch_bounds__(WAVES * 32)
mp_edge_kernel(const int* __restrict__ ei, const float* __restrict__ x,
               float* __restrict__ e, float* __restrict__ aggr, MLPW w, int E) {
  __shared__ _Float16 stg[WAVES][16 * 200];   // [16][192] input, pad 200
  __shared__ _Float16 act[WAVES][16 * 72];
  __shared__ int didx[WAVES][16];
  int lane = threadIdx.x & 31, wave = threadIdx.x >> 5;
  int tile = blockIdx.x * WAVES + wave;
  if (tile * 16 >= E) return;
  int row = lane & 15, hf = lane >> 4;
  int eid = tile * 16 + row;
  int s = ei[eid];          // edge_index[0] = src
  int d = ei[E + eid];      // edge_index[1] = dst
  if (hf == 0) didx[wave][row] = d;
  _Float16* st = stg[wave] + row * 200;
  stage32(x + (size_t)d * HID + hf * 32, st + hf * 32);         // x_i
  stage32(x + (size_t)s * HID + hf * 32, st + 64 + hf * 32);    // x_j
  stage32(e + (size_t)eid * HID + hf * 32, st + 128 + hf * 32); // e
  MEMFENCE();

  f8 c[4];
#pragma unroll
  for (int n = 0; n < 4; ++n) c[n] = (f8)0.f;
  gemm_lds<6>(stg[wave], 200, w.w0, lane, c);   // K=192
  bias_act(c, w.b0, lane, true);
  c_to_lds_h(c, act[wave], 72, lane);
#pragma unroll
  for (int n = 0; n < 4; ++n) c[n] = (f8)0.f;
  gemm_lds<2>(act[wave], 72, w.w1, lane, c);
  bias_act(c, w.b1, lane, true);
  c_to_lds_h(c, act[wave], 72, lane);
#pragma unroll
  for (int n = 0; n < 4; ++n) c[n] = (f8)0.f;
  gemm_lds<2>(act[wave], 72, w.w2, lane, c);
  bias_act(c, w.b2, lane, false);
  layernorm(c, w.gamma, w.beta, lane);          // c = m

  // e += m : bounce m through LDS (reuse stage buffer as f32, stride 100)
  float* mf = (float*)stg[wave];
  int col = lane & 15;
  MEMFENCE();
#pragma unroll
  for (int n = 0; n < 4; ++n)
#pragma unroll
    for (int r = 0; r < 8; ++r)
      mf[(r + hf * 8) * 100 + n * 16 + col] = c[n][r];
  MEMFENCE();
  const float* srcF = mf + row * 100 + hf * 32;
  float* erow = e + (size_t)eid * HID + hf * 32;
#pragma unroll
  for (int j = 0; j < 8; ++j) {
    float4 m4 = ((const float4*)srcF)[j];
    float4 o = ((const float4*)erow)[j];
    o.x += m4.x; o.y += m4.y; o.z += m4.z; o.w += m4.w;
    ((float4*)erow)[j] = o;
  }

  // segment-sum scatter: global_atomic_add_f32 into L2-resident aggr
  int dl[8];
#pragma unroll
  for (int r = 0; r < 8; ++r) dl[r] = didx[wave][r + hf * 8];
#pragma unroll
  for (int n = 0; n < 4; ++n)
#pragma unroll
    for (int r = 0; r < 8; ++r)
      unsafeAtomicAdd(aggr + (size_t)dl[r] * HID + n * 16 + col, c[n][r]);
}

// ---------------------------------------------------------------------------
// MP node pass: x = x + MLP([x | aggr]) (K=128).
// ---------------------------------------------------------------------------
__global__ void __launch_bounds__(WAVES * 32)
mp_node_kernel(float* __restrict__ x, const float* __restrict__ aggr, MLPW w,
               int N) {
  __shared__ _Float16 stg[WAVES][16 * 136];
  __shared__ _Float16 act[WAVES][16 * 72];
  int lane = threadIdx.x & 31, wave = threadIdx.x >> 5;
  int tile = blockIdx.x * WAVES + wave;
  if (tile * 16 >= N) return;
  int row = lane & 15, hf = lane >> 4;
  int grow = tile * 16 + row;
  _Float16* st = stg[wave] + row * 136;
  stage32(x + (size_t)grow * HID + hf * 32, st + hf * 32);
  stage32(aggr + (size_t)grow * HID + hf * 32, st + 64 + hf * 32);
  MEMFENCE();

  f8 c[4];
#pragma unroll
  for (int n = 0; n < 4; ++n) c[n] = (f8)0.f;
  gemm_lds<4>(stg[wave], 136, w.w0, lane, c);   // K=128
  bias_act(c, w.b0, lane, true);
  c_to_lds_h(c, act[wave], 72, lane);
#pragma unroll
  for (int n = 0; n < 4; ++n) c[n] = (f8)0.f;
  gemm_lds<2>(act[wave], 72, w.w1, lane, c);
  bias_act(c, w.b1, lane, true);
  c_to_lds_h(c, act[wave], 72, lane);
#pragma unroll
  for (int n = 0; n < 4; ++n) c[n] = (f8)0.f;
  gemm_lds<2>(act[wave], 72, w.w2, lane, c);
  bias_act(c, w.b2, lane, false);
  layernorm(c, w.gamma, w.beta, lane);

  // residual: x += u (LDS bounce for coalesced RMW)
  float* uf = (float*)stg[wave];   // stride 68 floats (136 halfs)
  int col = lane & 15;
  MEMFENCE();
#pragma unroll
  for (int n = 0; n < 4; ++n)
#pragma unroll
    for (int r = 0; r < 8; ++r)
      uf[(r + hf * 8) * 68 + n * 16 + col] = c[n][r];
  MEMFENCE();
  const float* srcF = uf + row * 68 + hf * 32;
  float* xrow = x + (size_t)grow * HID + hf * 32;
#pragma unroll
  for (int j = 0; j < 8; ++j) {
    float4 u4 = ((const float4*)srcF)[j];
    float4 o = ((const float4*)xrow)[j];
    o.x += u4.x; o.y += u4.y; o.z += u4.z; o.w += u4.w;
    ((float4*)xrow)[j] = o;
  }
}

// ---------------------------------------------------------------------------
// Decoder: out[n] = W2 . relu(W1 . relu(W0 . x + b0) + b1) + b2   (64->1)
// ---------------------------------------------------------------------------
__global__ void __launch_bounds__(WAVES * 32)
decode_kernel(const float* __restrict__ x, float* __restrict__ out,
              const _Float16* w0p, const _Float16* w1p, const float* b0,
              const float* b1, const float* w2, const float* b2, int N) {
  __shared__ _Float16 act[WAVES][16 * 72];
  int lane = threadIdx.x & 31, wave = threadIdx.x >> 5;
  int tile = blockIdx.x * WAVES + wave;
  if (tile * 16 >= N) return;
  int row = lane & 15, hf = lane >> 4;
  int grow = tile * 16 + row;
  stage32(x + (size_t)grow * HID + hf * 32, act[wave] + row * 72 + hf * 32);
  MEMFENCE();

  f8 c[4];
#pragma unroll
  for (int n = 0; n < 4; ++n) c[n] = (f8)0.f;
  gemm_lds<2>(act[wave], 72, w0p, lane, c);
  bias_act(c, b0, lane, true);
  c_to_lds_h(c, act[wave], 72, lane);
#pragma unroll
  for (int n = 0; n < 4; ++n) c[n] = (f8)0.f;
  gemm_lds<2>(act[wave], 72, w1p, lane, c);
  bias_act(c, b1, lane, true);

  // final 64->1: per-row dot with W2, cross-lane reduce within each half
  int col = lane & 15;
  float wv[4];
#pragma unroll
  for (int n = 0; n < 4; ++n) wv[n] = w2[n * 16 + col];
  float p[8];
#pragma unroll
  for (int r = 0; r < 8; ++r)
    p[r] = c[0][r] * wv[0] + c[1][r] * wv[1] + c[2][r] * wv[2] + c[3][r] * wv[3];
#pragma unroll
  for (int m = 1; m <= 8; m <<= 1)
#pragma unroll
    for (int r = 0; r < 8; ++r) p[r] += __shfl_xor(p[r], m, 32);
  if (col == 0) {
    float bb = b2[0];
#pragma unroll
    for (int r = 0; r < 8; ++r) out[tile * 16 + hf * 8 + r] = p[r] + bb;
  }
}

// ---------------------------------------------------------------------------
// Host launch.  d_in pytree order (jax sorts dict keys; 'W'<'b'<'beta'<'gamma'):
//  0 node_features, 1 edge_features, 2 edge_index[2,E] (int32),
//  3..10  edge_in  {W0,W1,W2,b0,b1,b2,beta,gamma}
//  11+16i mp[i].lin_edge {W0,W1,W2,b0,b1,b2,beta,gamma}
//  19+16i mp[i].lin_node {W0,W1,W2,b0,b1,b2,beta,gamma}
//  75..82 node_in {..}, 83..88 node_out {W0,W1,W2,b0,b1,b2}
// ---------------------------------------------------------------------------
extern "C" void kernel_launch(void* const* d_in, const int* in_sizes, int n_in,
                              void* d_out, int out_size, void* d_ws,
                              size_t ws_size, hipStream_t stream) {
  const int N = 100000, E = 1600000;
  const float* node_feat = (const float*)d_in[0];
  const float* edge_feat = (const float*)d_in[1];
  const int* ei = (const int*)d_in[2];
  auto F = [&](int i) { return (const float*)d_in[i]; };

  // workspace: [packed weights 0.5MB][x 25.6MB][aggr 25.6MB][e 409.6MB]
  char* ws = (char*)d_ws;
  _Float16* wp = (_Float16*)ws;
  float* x = (float*)(ws + (1 << 19));
  float* aggr = x + (size_t)N * HID;
  float* e = aggr + (size_t)N * HID;

  PackArgs pa;
  const float* wm[32];
  int kp[32], kr[32];
  wm[0] = F(75); kp[0] = 32;  kr[0] = 4;     // node_in.W0
  wm[1] = F(76); kp[1] = 64;  kr[1] = 64;
  wm[2] = F(77); kp[2] = 64;  kr[2] = 64;
  wm[3] = F(3);  kp[3] = 32;  kr[3] = 4;     // edge_in.W0
  wm[4] = F(4);  kp[4] = 64;  kr[4] = 64;
  wm[5] = F(5);  kp[5] = 64;  kr[5] = 64;
  wm[6] = F(83); kp[6] = 64;  kr[6] = 64;    // node_out.W0
  wm[7] = F(84); kp[7] = 64;  kr[7] = 64;    // node_out.W1
  for (int i = 0; i < 4; ++i) {
    int b = 11 + 16 * i, m = 8 + 6 * i;
    wm[m + 0] = F(b + 0);  kp[m + 0] = 192; kr[m + 0] = 192;  // lin_edge.W0
    wm[m + 1] = F(b + 1);  kp[m + 1] = 64;  kr[m + 1] = 64;
    wm[m + 2] = F(b + 2);  kp[m + 2] = 64;  kr[m + 2] = 64;
    wm[m + 3] = F(b + 8);  kp[m + 3] = 128; kr[m + 3] = 128;  // lin_node.W0
    wm[m + 4] = F(b + 9);  kp[m + 4] = 64;  kr[m + 4] = 64;
    wm[m + 5] = F(b + 10); kp[m + 5] = 64;  kr[m + 5] = 64;
  }
  int cum = 0, totalFrags = 0;
  for (int m = 0; m < 32; ++m) {
    pa.W[m] = wm[m]; pa.Kpad[m] = kp[m]; pa.Kreal[m] = kr[m]; pa.off[m] = cum;
    int nf = (kp[m] >> 5) * 4;
    totalFrags += nf;
    cum += nf * 512;
  }
  pack_kernel<<<totalFrags, 32, 0, stream>>>(pa, wp);

  MLPW encN = {wp + pa.off[0], wp + pa.off[1], wp + pa.off[2],
               F(78), F(79), F(80), F(82), F(81)};
  MLPW encE = {wp + pa.off[3], wp + pa.off[4], wp + pa.off[5],
               F(6), F(7), F(8), F(10), F(9)};
  int nBlocksN = (N / 16 + WAVES - 1) / WAVES;
  int nBlocksE = (E / 16) / WAVES;
  encoder_kernel<<<nBlocksN, WAVES * 32, 0, stream>>>(node_feat, x, encN, N);
  encoder_kernel<<<nBlocksE, WAVES * 32, 0, stream>>>(edge_feat, e, encE, E);

  for (int i = 0; i < 4; ++i) {
    int b = 11 + 16 * i, m = 8 + 6 * i;
    MLPW mpe = {wp + pa.off[m], wp + pa.off[m + 1], wp + pa.off[m + 2],
                F(b + 3), F(b + 4), F(b + 5), F(b + 7), F(b + 6)};
    MLPW mpn = {wp + pa.off[m + 3], wp + pa.off[m + 4], wp + pa.off[m + 5],
                F(b + 11), F(b + 12), F(b + 13), F(b + 15), F(b + 14)};
    hipMemsetAsync(aggr, 0, (size_t)N * HID * sizeof(float), stream);
    mp_edge_kernel<<<nBlocksE, WAVES * 32, 0, stream>>>(ei, x, e, aggr, mpe, E);
    mp_node_kernel<<<nBlocksN, WAVES * 32, 0, stream>>>(x, aggr, mpn, N);
  }
  decode_kernel<<<nBlocksN, WAVES * 32, 0, stream>>>(
      x, (float*)d_out, wp + pa.off[6], wp + pa.off[7], F(86), F(87), F(85),
      F(88), N);
}
